// WindowedAttn_14645838479939
// MI455X (gfx1250) — matrix operand order repaired and checked by reading, not compiled, not run on hardware
//
#include <hip/hip_runtime.h>
#include <stdint.h>

typedef __attribute__((ext_vector_type(16))) __bf16 v16bf;
typedef __attribute__((ext_vector_type(8)))  float  v8f;
typedef unsigned short u16;

#define D_MODEL 1024
#define N_HEADS 16
#define D_HEAD  64
#define WINDOW  256
#define BATCH   2
#define SEQ     2048
#define M_TOT   (BATCH * SEQ)          // 4096 rows total
#define SCALE_F 0.125f                 // 64^-0.5

// ---------- helpers ----------

__device__ __forceinline__ u16 f2bf(float f) {
  uint32_t u = __float_as_uint(f);
  u += 0x7FFFu + ((u >> 16) & 1u);     // round-to-nearest-even
  return (u16)(u >> 16);
}

union FragAB {
  v16bf v;
  uint4 q[2];
  u16   h[16];
};

// A fragment: 16x32 (MxK), src row-major, row stride ld (elements).
// lanes 0-15: row = row0+l, elems 0-15 = K k0..k0+15
// lanes 16-31: same row, K k0+16..k0+31
__device__ __forceinline__ v16bf load_a(const u16* __restrict__ p, int ld,
                                        int row0, int k0, int lane) {
  const int half = lane >> 4, l = lane & 15;
  const u16* r = p + (size_t)(row0 + l) * ld;
  FragAB f;
  f.q[0] = *(const uint4*)(r + k0 + half * 16);
  f.q[1] = *(const uint4*)(r + k0 + half * 16 + 8);
  return f.v;
}

// B fragment: 32x16 (KxN) read from Bt stored [N,K] row-major
// (column n of B == row n of Bt). lanes 0-15: col n0+l, K k0..k0+15;
// lanes 16-31: col n0+l, K k0+16..k0+31  (contiguous 16-element runs)
__device__ __forceinline__ v16bf load_b(const u16* __restrict__ p, int ld,
                                        int n0, int k0, int lane) {
  const int half = lane >> 4, l = lane & 15;
  const u16* r = p + (size_t)(n0 + l) * ld + k0 + half * 16;
  FragAB f;
  f.q[0] = *(const uint4*)(r);
  f.q[1] = *(const uint4*)(r + 8);
  return f.v;
}

// B fragment with clamped source row (keeps loads in-bounds; masked anyway)
__device__ __forceinline__ v16bf load_b_rowclamp(const u16* __restrict__ p, int ld,
                                                 int n0, int k0, int lane, int maxrow) {
  const int half = lane >> 4, l = lane & 15;
  int row = n0 + l; row = row < maxrow ? row : maxrow;
  const u16* r = p + (size_t)row * ld + k0 + half * 16;
  FragAB f;
  f.q[0] = *(const uint4*)(r);
  f.q[1] = *(const uint4*)(r + 8);
  return f.v;
}

__device__ __forceinline__ v8f wmma_bf16(v16bf a, v16bf b, v8f c) {
  return __builtin_amdgcn_wmma_f32_16x16x32_bf16(
      /*neg_a=*/false, a, /*neg_b=*/false, b,
      /*c_mod=*/(short)0, c, /*reuse_a=*/false, /*reuse_b=*/false);
}

// ---------- cast / transpose kernels ----------

__global__ void cast_f32_bf16(const float* __restrict__ in, u16* __restrict__ out, int n) {
  int i = blockIdx.x * 256 + threadIdx.x;
  if (i < n) out[i] = f2bf(in[i]);
}

// in: [K,N] fp32 row-major -> out: [N,K] bf16 row-major
__global__ void transpose_cast(const float* __restrict__ in, u16* __restrict__ out,
                               int K, int N) {
  int i = blockIdx.x * 256 + threadIdx.x;
  if (i < K * N) {
    int n = i / K, k = i - n * K;
    out[i] = f2bf(in[(size_t)k * N + n]);
  }
}

// ---------- GEMM 1: qkv = x @ W_qkv + b, scattered into Q,K (head-major) and V^T ----------

__global__ void __launch_bounds__(128) gemm_qkv(
    const u16* __restrict__ A,      // x_bf16  [4096,1024]
    const u16* __restrict__ Bt,     // WqkvT   [3072,1024]
    const float* __restrict__ bias, // [3072]
    u16* __restrict__ Qo, u16* __restrict__ Ko, u16* __restrict__ Vt) {
  const int lane = threadIdx.x & 31;
  const int wid  = blockIdx.x * 4 + (threadIdx.x >> 5);
  const int ntiles = (3 * D_MODEL) / 64;            // 48
  const int m0 = (wid / ntiles) * 16;
  const int n0 = (wid % ntiles) * 64;
  const int half = lane >> 4, l = lane & 15;

  v8f acc0 = {}, acc1 = {}, acc2 = {}, acc3 = {};
  for (int k0 = 0; k0 < D_MODEL; k0 += 32) {
    __builtin_prefetch((const void*)(A + (size_t)(m0 + l) * D_MODEL + k0 + 256), 0, 1);
    __builtin_prefetch((const void*)(Bt + (size_t)(n0 + l) * D_MODEL + k0 + 256), 0, 1);
    v16bf a  = load_a(A, D_MODEL, m0, k0, lane);
    v16bf b0 = load_b(Bt, D_MODEL, n0 +  0, k0, lane);
    v16bf b1 = load_b(Bt, D_MODEL, n0 + 16, k0, lane);
    v16bf b2 = load_b(Bt, D_MODEL, n0 + 32, k0, lane);
    v16bf b3 = load_b(Bt, D_MODEL, n0 + 48, k0, lane);
    acc0 = wmma_bf16(a, b0, acc0);
    acc1 = wmma_bf16(a, b1, acc1);
    acc2 = wmma_bf16(a, b2, acc2);
    acc3 = wmma_bf16(a, b3, acc3);
  }

  v8f accs[4] = {acc0, acc1, acc2, acc3};
#pragma unroll
  for (int t = 0; t < 4; ++t) {
    const int c = n0 + t * 16 + l;                  // global qkv column
    const float bv = bias[c];
    const int which = c >> 10;                      // 0=q 1=k 2=v
    const int cc = c & (D_MODEL - 1);
    const int h = cc >> 6, d = cc & 63;
#pragma unroll
    for (int r = 0; r < 8; ++r) {
      const int row = m0 + r + half * 8;
      const int b  = row >> 11;                     // batch
      const int s  = row & (SEQ - 1);
      const u16 val = f2bf(accs[t][r] + bv);
      const size_t bh = (size_t)(b * N_HEADS + h);
      if (which == 0)      Qo[(bh * SEQ + s) * D_HEAD + d] = val;
      else if (which == 1) Ko[(bh * SEQ + s) * D_HEAD + d] = val;
      else                 Vt[(bh * D_HEAD + d) * SEQ + s] = val;   // transposed V
    }
  }
}

// ---------- flash-style windowed attention (anti-causal window: i <= j <= i+255) ----------

__global__ void __launch_bounds__(128) attn_kernel(
    const u16* __restrict__ Q, const u16* __restrict__ K, const u16* __restrict__ Vt,
    u16* __restrict__ Out /* [B,S,D_MODEL] bf16 */) {
  __shared__ __align__(16) u16 lds[4][16 * 32];     // per-wave 16x32 prob tile

  const int lane = threadIdx.x & 31;
  const int wv   = threadIdx.x >> 5;
  const int wid  = blockIdx.x * 4 + wv;
  const int ib = wid & (SEQ / 16 - 1);              // 0..127
  const int h  = (wid >> 7) & (N_HEADS - 1);
  const int b  = wid >> 11;
  const int i0 = ib * 16;
  const int half = lane >> 4, l = lane & 15;

  const u16* qb = Q  + (size_t)(b * N_HEADS + h) * SEQ * D_HEAD;
  const u16* kb = K  + (size_t)(b * N_HEADS + h) * SEQ * D_HEAD;
  const u16* vb = Vt + (size_t)(b * N_HEADS + h) * D_HEAD * SEQ;

  const v16bf qa0 = load_a(qb, D_HEAD, i0, 0, lane);
  const v16bf qa1 = load_a(qb, D_HEAD, i0, 32, lane);

  float m[8], lsum[8];
#pragma unroll
  for (int r = 0; r < 8; ++r) { m[r] = -1e30f; lsum[r] = 0.f; }
  v8f o0 = {}, o1 = {}, o2 = {}, o3 = {};

  for (int t = 0; t < 9; ++t) {                     // strips of 32 keys
    const int j0 = i0 + t * 32;

    // scores: Q(16x64) @ K^T(64x32) -> two 16x16 tiles
    v8f s0 = {}, s1 = {};
    {
      v16bf kb00 = load_b_rowclamp(kb, D_HEAD, j0,      0,  lane, SEQ - 1);
      v16bf kb01 = load_b_rowclamp(kb, D_HEAD, j0,      32, lane, SEQ - 1);
      v16bf kb10 = load_b_rowclamp(kb, D_HEAD, j0 + 16, 0,  lane, SEQ - 1);
      v16bf kb11 = load_b_rowclamp(kb, D_HEAD, j0 + 16, 32, lane, SEQ - 1);
      s0 = wmma_bf16(qa0, kb00, s0);
      s0 = wmma_bf16(qa1, kb01, s0);
      s1 = wmma_bf16(qa0, kb10, s1);
      s1 = wmma_bf16(qa1, kb11, s1);
    }

    // scale + window mask + per-row strip max
    float sm[8];
#pragma unroll
    for (int r = 0; r < 8; ++r) {
      const int row = i0 + r + half * 8;
      const int jA = j0 + l, jB = j0 + 16 + l;
      float a0 = s0[r] * SCALE_F, a1 = s1[r] * SCALE_F;
      const bool okA = (row <= jA) && (jA <= row + WINDOW - 1) && (jA < SEQ);
      const bool okB = (row <= jB) && (jB <= row + WINDOW - 1) && (jB < SEQ);
      a0 = okA ? a0 : -1e30f;
      a1 = okB ? a1 : -1e30f;
      s0[r] = a0; s1[r] = a1;
      sm[r] = fmaxf(a0, a1);
    }
#pragma unroll
    for (int off = 8; off >= 1; off >>= 1)
#pragma unroll
      for (int r = 0; r < 8; ++r)
        sm[r] = fmaxf(sm[r], __shfl_xor(sm[r], off, 32));

    float alpha[8];
#pragma unroll
    for (int r = 0; r < 8; ++r) {
      const float mn = fmaxf(m[r], sm[r]);
      alpha[r] = __expf(m[r] - mn);
      m[r] = mn;
    }

    // probabilities; stash bf16 tile in LDS in row-major for A-layout reload
    float rs[8];
#pragma unroll
    for (int r = 0; r < 8; ++r) {
      const float p0 = __expf(s0[r] - m[r]);
      const float p1 = __expf(s1[r] - m[r]);
      rs[r] = p0 + p1;
      const int row = r + half * 8;
      lds[wv][row * 32 + l]      = f2bf(p0);
      lds[wv][row * 32 + 16 + l] = f2bf(p1);
    }
#pragma unroll
    for (int off = 8; off >= 1; off >>= 1)
#pragma unroll
      for (int r = 0; r < 8; ++r)
        rs[r] += __shfl_xor(rs[r], off, 32);
#pragma unroll
    for (int r = 0; r < 8; ++r) {
      lsum[r] = lsum[r] * alpha[r] + rs[r];
      o0[r] *= alpha[r]; o1[r] *= alpha[r]; o2[r] *= alpha[r]; o3[r] *= alpha[r];
    }

    asm volatile("s_wait_dscnt 0" ::: "memory");    // DS stores visible before reload

    // reload P as A-fragment (16x32)
    FragAB pf;
    {
      const u16* lr = &lds[wv][l * 32];
      pf.q[0] = *(const uint4*)(lr + half * 16);
      pf.q[1] = *(const uint4*)(lr + half * 16 + 8);
    }

    // O += P(16x32) @ V^T-slices(32x16) for 4 d-blocks
#pragma unroll
    for (int db = 0; db < 4; ++db) {
      const int d = db * 16 + l;
      int js = j0 + half * 16;
      js = js < (SEQ - 16) ? js : (SEQ - 16);       // clamp (masked probs are 0)
      const u16* vr = vb + (size_t)d * SEQ + js;
      FragAB vf;
      vf.q[0] = *(const uint4*)(vr);
      vf.q[1] = *(const uint4*)(vr + 8);
      if (db == 0)      o0 = wmma_bf16(pf.v, vf.v, o0);
      else if (db == 1) o1 = wmma_bf16(pf.v, vf.v, o1);
      else if (db == 2) o2 = wmma_bf16(pf.v, vf.v, o2);
      else              o3 = wmma_bf16(pf.v, vf.v, o3);
    }
  }

  // normalize and store to [B,S,D_MODEL] bf16 (heads interleaved back)
  float inv[8];
#pragma unroll
  for (int r = 0; r < 8; ++r) inv[r] = 1.0f / lsum[r];
  v8f os[4] = {o0, o1, o2, o3};
  u16* ob = Out + (size_t)b * SEQ * D_MODEL + (size_t)h * D_HEAD;
#pragma unroll
  for (int db = 0; db < 4; ++db) {
    const int d = db * 16 + l;
#pragma unroll
    for (int r = 0; r < 8; ++r) {
      const int row = i0 + r + half * 8;
      ob[(size_t)row * D_MODEL + d] = f2bf(os[db][r] * inv[r]);
    }
  }
}

// ---------- GEMM 2: out = attn_out @ W_out + b_out (fp32 result) ----------

__global__ void __launch_bounds__(128) gemm_out(
    const u16* __restrict__ A,      // attn_out bf16 [4096,1024]
    const u16* __restrict__ Bt,     // WoutT bf16    [1024,1024]
    const float* __restrict__ bias, // [1024]
    float* __restrict__ C) {        // [4096,1024] fp32
  const int lane = threadIdx.x & 31;
  const int wid  = blockIdx.x * 4 + (threadIdx.x >> 5);
  const int ntiles = D_MODEL / 64;                  // 16
  const int m0 = (wid / ntiles) * 16;
  const int n0 = (wid % ntiles) * 64;
  const int half = lane >> 4, l = lane & 15;

  v8f acc0 = {}, acc1 = {}, acc2 = {}, acc3 = {};
  for (int k0 = 0; k0 < D_MODEL; k0 += 32) {
    __builtin_prefetch((const void*)(A + (size_t)(m0 + l) * D_MODEL + k0 + 256), 0, 1);
    v16bf a  = load_a(A, D_MODEL, m0, k0, lane);
    v16bf b0 = load_b(Bt, D_MODEL, n0 +  0, k0, lane);
    v16bf b1 = load_b(Bt, D_MODEL, n0 + 16, k0, lane);
    v16bf b2 = load_b(Bt, D_MODEL, n0 + 32, k0, lane);
    v16bf b3 = load_b(Bt, D_MODEL, n0 + 48, k0, lane);
    acc0 = wmma_bf16(a, b0, acc0);
    acc1 = wmma_bf16(a, b1, acc1);
    acc2 = wmma_bf16(a, b2, acc2);
    acc3 = wmma_bf16(a, b3, acc3);
  }

  v8f accs[4] = {acc0, acc1, acc2, acc3};
#pragma unroll
  for (int t = 0; t < 4; ++t) {
    const int c = n0 + t * 16 + l;
    const float bv = bias[c];
#pragma unroll
    for (int r = 0; r < 8; ++r) {
      const int row = m0 + r + half * 8;
      C[(size_t)row * D_MODEL + c] = accs[t][r] + bv;
    }
  }
}

// ---------- launcher ----------

extern "C" void kernel_launch(void* const* d_in, const int* in_sizes, int n_in,
                              void* d_out, int out_size, void* d_ws, size_t ws_size,
                              hipStream_t stream) {
  const float* x     = (const float*)d_in[0];
  const float* W_qkv = (const float*)d_in[1];
  const float* b_qkv = (const float*)d_in[2];
  const float* W_out = (const float*)d_in[3];
  const float* b_out = (const float*)d_in[4];
  float* out = (float*)d_out;

  char* ws = (char*)d_ws;
  u16* x_bf  = (u16*)ws; ws += (size_t)M_TOT * D_MODEL * 2;            // 8 MB
  u16* wqkvT = (u16*)ws; ws += (size_t)3 * D_MODEL * D_MODEL * 2;      // 6 MB
  u16* woutT = (u16*)ws; ws += (size_t)D_MODEL * D_MODEL * 2;          // 2 MB
  u16* Qb    = (u16*)ws; ws += (size_t)M_TOT * D_MODEL * 2;            // 8 MB
  u16* Kb    = (u16*)ws; ws += (size_t)M_TOT * D_MODEL * 2;            // 8 MB
  u16* Vtb   = (u16*)ws; ws += (size_t)M_TOT * D_MODEL * 2;            // 8 MB
  u16* AOut  = (u16*)ws; ws += (size_t)M_TOT * D_MODEL * 2;            // 8 MB

  const int nX = M_TOT * D_MODEL;
  cast_f32_bf16<<<(nX + 255) / 256, 256, 0, stream>>>(x, x_bf, nX);
  transpose_cast<<<(3 * D_MODEL * D_MODEL + 255) / 256, 256, 0, stream>>>(
      W_qkv, wqkvT, D_MODEL, 3 * D_MODEL);
  transpose_cast<<<(D_MODEL * D_MODEL + 255) / 256, 256, 0, stream>>>(
      W_out, woutT, D_MODEL, D_MODEL);

  // (M_TOT/16) * (3*D_MODEL/64) = 256*48 = 12288 waves, 4 waves/block
  gemm_qkv<<<12288 / 4, 128, 0, stream>>>(x_bf, wqkvT, b_qkv, Qb, Kb, Vtb);

  // B*H*(S/16) = 4096 waves
  attn_kernel<<<4096 / 4, 128, 0, stream>>>(Qb, Kb, Vtb, AOut);

  // 256*16 = 4096 waves
  gemm_out<<<4096 / 4, 128, 0, stream>>>(AOut, woutT, b_out, out);
}
